// Decoder_41154376630800
// MI455X (gfx1250) — compile-verified, matrix operand order checked by reading, not run on hardware
//
#include <hip/hip_runtime.h>
#include <hip/hip_bf16.h>

// Problem constants (match reference)
#define BB 4
#define TT 1024
#define VV 32000
#define DD 1024
#define HH 16
#define DKK 64
#define MROWS (BB * TT)          // 4096

typedef __attribute__((ext_vector_type(16))) _Float16 v16h;
typedef __attribute__((ext_vector_type(8)))  float    v8f;
typedef unsigned int u32x4 __attribute__((ext_vector_type(4)));
typedef int          i32x8 __attribute__((ext_vector_type(8)));
typedef int          i32x4 __attribute__((ext_vector_type(4)));

#if defined(__has_builtin)
#if __has_builtin(__builtin_amdgcn_tensor_load_to_lds)
#define HAVE_TDM 1
#endif
#endif

// ---- WMMA helpers ---------------------------------------------------------

__device__ __forceinline__ v8f wmma16(v16h a, v16h b, v8f c) {
  // (neg_a, A, neg_b, B, c_mod, C, reuse_a, reuse_b)
  return __builtin_amdgcn_wmma_f32_16x16x32_f16(false, a, false, b, (short)0, c,
                                                false, false);
}

// ISA 7.12.2: 16-bit A-matrix 16x32 VGPR layout.
// lane half hh (0/1) and vgpr pair j -> K offset
__device__ __forceinline__ int frag_koff(int j, int hh) {
  return ((j >> 2) << 4) + (hh << 3) + ((j & 3) << 1); // (j/4)*16 + hh*8 + (j%4)*2
}

// Fragment (16 rows x 32 k) from row-major f16 with stride (halves).
// Contiguous K-pairs -> compiler merges into ds_load_b128 pairs.
__device__ __forceinline__ v16h load_frag_rm(const _Float16* base, int stride,
                                             int row_l, int hh) {
  v16h a;
  const _Float16* rp = base + (size_t)row_l * stride;
#pragma unroll
  for (int j = 0; j < 8; ++j) {
    int k = frag_koff(j, hh);
    a[2 * j]     = rp[k];
    a[2 * j + 1] = rp[k + 1];
  }
  return a;
}

// ---- TDM (Tensor Data Mover) 2D tile load, D# per ISA cdna5 ch.8 ----------
// Loads a tile_d1 x tile_d0 (elements, 2B each) tile whose first element is at
// gptr, with row stride stride_elems, into LDS at lds_off with padding
// (pad codes per D# group1: interval 0=2dw..7=256dw, amount code = dwords-1).

__device__ __forceinline__ unsigned lds_offset(const void* p) {
  return (unsigned)(size_t)p;   // low 32 bits of a generic LDS pointer
}

#ifdef HAVE_TDM
__device__ __forceinline__ void tdm_load_2d_f16(unsigned lds_off,
                                                const void* gptr,
                                                unsigned tensor_d0,
                                                unsigned tensor_d1,
                                                unsigned tile_d0,
                                                unsigned tile_d1,
                                                unsigned stride_elems,
                                                unsigned pad_interval_code,
                                                unsigned pad_amount_code) {
  unsigned long long ga = (unsigned long long)(size_t)gptr;
  u32x4 g0;
  g0[0] = 1u;                                   // count=1, user mode
  g0[1] = lds_off;                              // lds_addr
  g0[2] = (unsigned)ga;                         // global_addr[31:0]
  g0[3] = ((unsigned)(ga >> 32) & 0x01FFFFFFu)  // global_addr[56:32]
          | (2u << 30);                         // type = 2 ("image")
  i32x8 g1;
  g1[0] = (int)((1u << 16)                      // data_size = 2 bytes
                | (1u << 20)                    // pad_enable
                | (pad_interval_code << 22)
                | (pad_amount_code << 25));
  g1[1] = (int)((tensor_d0 & 0xFFFFu) << 16);               // tensor_dim0 lo16
  g1[2] = (int)((tensor_d0 >> 16) | ((tensor_d1 & 0xFFFFu) << 16));
  g1[3] = (int)((tensor_d1 >> 16) | (tile_d0 << 16));       // tile_dim0
  g1[4] = (int)tile_d1;                                     // tile_dim1 (dim2=0)
  g1[5] = (int)stride_elems;                                // dim0_stride[31:0]
  g1[6] = 0;                                                // dim0_stride[47:32]
  g1[7] = 0;
  // clang-23 / therock-10.0 headers: 6-arg form (g0, g1, g2, g3, g4, cpol)
  __builtin_amdgcn_tensor_load_to_lds(g0, g1, (i32x4)0, (i32x4)0, (i32x8)0, 0);
}
#endif

// ---- 1. converts / embedding ----------------------------------------------

__global__ void f32_to_f16_kernel(const float* __restrict__ in,
                                  _Float16* __restrict__ out, long n) {
  long i = (long)blockIdx.x * blockDim.x + threadIdx.x;
  long stride = (long)gridDim.x * blockDim.x;
  for (; i < n; i += stride) out[i] = (_Float16)in[i];
}

__global__ void embed_kernel(const int* __restrict__ inputs,
                             const float* __restrict__ tok,
                             const float* __restrict__ pos,
                             _Float16* __restrict__ xh) {
  int row = blockIdx.x;            // 0..B*T-1
  int t = row & (TT - 1);
  int v = inputs[row];
  const float* tp = tok + (size_t)v * DD;
  const float* pp = pos + (size_t)t * DD;
  _Float16* xp = xh + (size_t)row * DD;
  for (int d = threadIdx.x; d < DD; d += blockDim.x)
    xp[d] = (_Float16)(tp[d] * 32.0f + pp[d]);   // sqrt(1024) = 32
}

// ---- 2. QKV projection GEMM (M=4096, N=1024, K=1024) ----------------------
// A: xh [M][K] f16 row-major.  W: f16 in [H][D][DK] layout.  out: f16 [B][H][T][DK]

#define BM 64
#define BN 128
#define BK 32
#define SA 40     // As LDS row stride (halves): 32 + 8 pad, 16B-aligned rows
#define SBT 40    // BsT (transposed [n][k]) LDS row stride (halves)

__global__ __launch_bounds__(256)
void gemm_qkv_kernel(const _Float16* __restrict__ A,
                     const _Float16* __restrict__ W,
                     const float* __restrict__ bias,       // [H][DK]
                     _Float16* __restrict__ out) {
  __shared__ _Float16 As[BM * SA];        // 64 x 40
  __shared__ _Float16 BsT[BN * SBT];      // 128 x 40, [n][k]
  const int tid = threadIdx.x;
  const int lane = tid & 31, w = tid >> 5;
  const int hh = lane >> 4, nl = lane & 15, ml = lane & 15;
  const int m0 = blockIdx.y * BM;
  const int n0 = blockIdx.x * BN;
  const int wm = w & 3;      // m sub-tile (16 rows)
  const int wn = w >> 2;     // column half (64 cols)
  v8f acc[4] = {};

  for (int k0 = 0; k0 < DD; k0 += BK) {
    __syncthreads();
#ifdef HAVE_TDM
    if (w == 0)   // wave-level DMA: A tile 64x32 halves, LDS stride 40
      tdm_load_2d_f16(lds_offset(As), A + (size_t)m0 * DD + k0,
                      DD, MROWS, BK, BM, DD, /*interval 16dw*/3, /*4dw*/3);
#else
    {   // A tile: 64x32 halves. thread: row=tid/4, 8 halves (16B)
      int r = tid >> 2, p = (tid & 3) * 8;
      *(float4*)(As + r * SA + p) =
          *(const float4*)(A + (size_t)(m0 + r) * DD + k0 + p);
    }
#endif
    {   // B tile 32x128 from [H][D][DK]; store transposed [n][k] into LDS
      int kr = tid >> 3, p = (tid & 7) * 16;
      int n = n0 + p;
      int hN = n >> 6, dk = n & 63;     // 16-half chunk stays in one head
      const float4* src =
          (const float4*)(W + ((size_t)hN * DD + (k0 + kr)) * DKK + dk);
      float4 v0 = src[0], v1 = src[1];
      _Float16 tmp[16];
      *(float4*)&tmp[0] = v0;
      *(float4*)&tmp[8] = v1;
#pragma unroll
      for (int i = 0; i < 16; ++i) BsT[(p + i) * SBT + kr] = tmp[i];
    }
#ifdef HAVE_TDM
    if (w == 0) __builtin_amdgcn_s_wait_tensorcnt(0);
#endif
    __syncthreads();
    v16h a = load_frag_rm(As + wm * 16 * SA, SA, ml, hh);
#pragma unroll
    for (int nt = 0; nt < 4; ++nt) {
      v16h bf = load_frag_rm(BsT + (wn * 64 + nt * 16) * SBT, SBT, nl, hh);
      acc[nt] = wmma16(a, bf, acc[nt]);
    }
  }
#pragma unroll
  for (int nt = 0; nt < 4; ++nt) {
    int n = n0 + wn * 64 + nt * 16 + nl;
    int hN = n >> 6, dk = n & 63;
    float bv = bias[hN * DKK + dk];
#pragma unroll
    for (int r = 0; r < 8; ++r) {
      int m = m0 + wm * 16 + r + 8 * hh;         // global row = b*T + t
      int b = m >> 10, t = m & (TT - 1);
      out[(((size_t)b * HH + hN) * TT + t) * DKK + dk] =
          (_Float16)(acc[nt][r] + bv);
    }
  }
}

// ---- 3. attention: one block per (b, h, 64-query tile) --------------------
// Full 64x1024 score stripe in LDS (MI455X: 320KB/workgroup).

#define SP 1028   // padded row stride (floats) to dodge bank conflicts

__global__ __launch_bounds__(256)
void attn_kernel(const _Float16* __restrict__ qh,
                 const _Float16* __restrict__ kh,
                 const _Float16* __restrict__ vh,
                 _Float16* __restrict__ oh) {
  extern __shared__ float smem[];
  float* S      = smem;               // 64 * SP
  float* red    = S + 64 * SP;        // 256
  float* rowmax = red + 256;          // 64
  float* rowsum = rowmax + 64;        // 64

  const int tid = threadIdx.x, lane = tid & 31, w = tid >> 5;
  const int hh = lane >> 4, nl = lane & 15, ml = lane & 15;
  const int qt = blockIdx.x;          // query tile 0..15
  const int h  = blockIdx.y;
  const int b  = blockIdx.z;
  const int t0 = qt * 64;
  const size_t headBase = ((size_t)b * HH + h) * TT * DKK;   // [T][DK] halves

  // -------- scores = Q K^T, mask, scale --------
  const int wm  = w & 3;     // m-tile of the 64 queries
  const int par = w >> 2;    // n-tile parity
  v16h qa[2];
#pragma unroll
  for (int ks = 0; ks < 2; ++ks) {
#pragma unroll
    for (int j = 0; j < 8; ++j) {
      int k = ks * 32 + frag_koff(j, hh);
      const _Float16* qp =
          qh + headBase + (size_t)(t0 + wm * 16 + ml) * DKK + k;
      qa[ks][2 * j]     = qp[0];
      qa[ks][2 * j + 1] = qp[1];
    }
  }
  for (int nt = par; nt < TT / 16; nt += 2) {
    v8f acc = {};
#pragma unroll
    for (int ks = 0; ks < 2; ++ks) {
      v16h bf;   // B = K^T (DK x T): element (k=dk, n=s) = kh[s*64 + dk]
#pragma unroll
      for (int j = 0; j < 8; ++j) {
        int k = ks * 32 + frag_koff(j, hh);
        const _Float16* kp =
            kh + headBase + (size_t)(nt * 16 + nl) * DKK + k;
        bf[2 * j]     = kp[0];
        bf[2 * j + 1] = kp[1];
      }
      acc = wmma16(qa[ks], bf, acc);
    }
#pragma unroll
    for (int r = 0; r < 8; ++r) {
      int mq = wm * 16 + r + 8 * hh;
      int t = t0 + mq;
      int s = nt * 16 + nl;
      float val = (s > t) ? -1e9f : acc[r];     // mask BEFORE the 1/sqrt(dk)
      S[mq * SP + s] = val * 0.125f;
    }
  }
  __syncthreads();

  // -------- row softmax over 1024 keys (4 threads per row) --------
  {
    int row = tid >> 2, part = tid & 3;
    float* rp = S + row * SP + part * 256;
    float mx = -3.0e38f;
    for (int i = 0; i < 256; ++i) mx = fmaxf(mx, rp[i]);
    red[tid] = mx;
    __syncthreads();
    if (part == 0)
      rowmax[row] = fmaxf(fmaxf(red[row * 4], red[row * 4 + 1]),
                          fmaxf(red[row * 4 + 2], red[row * 4 + 3]));
    __syncthreads();
    float m = rowmax[row];
    float sum = 0.f;
    for (int i = 0; i < 256; ++i) {
      float e = __expf(rp[i] - m);
      rp[i] = e;
      sum += e;
    }
    red[tid] = sum;
    __syncthreads();
    if (part == 0)
      rowsum[row] = red[row * 4] + red[row * 4 + 1] +
                    red[row * 4 + 2] + red[row * 4 + 3];
    __syncthreads();
    float rinv = 1.0f / rowsum[row];
    for (int i = 0; i < 256; ++i) rp[i] *= rinv;
  }
  __syncthreads();

  // -------- out = P (64x1024) x V (1024x64) --------
#pragma unroll
  for (int tt = 0; tt < 2; ++tt) {
    int nt = (w >> 2) + tt * 2;        // 0..3
    v8f acc = {};
    for (int kc = 0; kc < TT; kc += 32) {
      v16h a, bf;
#pragma unroll
      for (int j = 0; j < 8; ++j) {
        int k = frag_koff(j, hh);
        const float* sp = S + (size_t)(wm * 16 + ml) * SP + kc + k;
        a[2 * j]     = (_Float16)sp[0];
        a[2 * j + 1] = (_Float16)sp[1];
        const _Float16* vp =
            vh + headBase + (size_t)(kc + k) * DKK + nt * 16 + nl;
        bf[2 * j]     = vp[0];
        bf[2 * j + 1] = vp[DKK];
      }
      acc = wmma16(a, bf, acc);
    }
#pragma unroll
    for (int r = 0; r < 8; ++r) {
      int m = wm * 16 + r + 8 * hh;
      oh[((size_t)(b * TT + t0 + m)) * DD + h * DKK + nt * 16 + nl] =
          (_Float16)acc[r];
    }
  }
}

// ---- 4. logits GEMM (M=4096, N=32000, K=1024) + bias ----------------------

__global__ __launch_bounds__(256)
void gemm_logits_kernel(const _Float16* __restrict__ A,    // oh [M][K]
                        const _Float16* __restrict__ Wo,   // f16 [K][N]
                        const float* __restrict__ bo,      // [N]
                        float* __restrict__ out) {         // [M][N]
  __shared__ _Float16 As[BM * SA];
  __shared__ _Float16 BsT[BN * SBT];      // transposed [n][k]
  const int tid = threadIdx.x;
  const int lane = tid & 31, w = tid >> 5;
  const int hh = lane >> 4, nl = lane & 15, ml = lane & 15;
  const int m0 = blockIdx.y * BM;
  const int n0 = blockIdx.x * BN;
  const int wm = w & 3;
  const int wn = w >> 2;
  v8f acc[4] = {};

  for (int k0 = 0; k0 < DD; k0 += BK) {
    __syncthreads();
#ifdef HAVE_TDM
    if (w == 0)
      tdm_load_2d_f16(lds_offset(As), A + (size_t)m0 * DD + k0,
                      DD, MROWS, BK, BM, DD, /*interval 16dw*/3, /*4dw*/3);
#else
    {
      int r = tid >> 2, p = (tid & 3) * 8;
      *(float4*)(As + r * SA + p) =
          *(const float4*)(A + (size_t)(m0 + r) * DD + k0 + p);
    }
#endif
    {   // B tile 32x128 (row-major [K][V]) -> transposed LDS [n][k]
      int kr = tid >> 3, p = (tid & 7) * 16;
      const float4* src =
          (const float4*)(Wo + (size_t)(k0 + kr) * VV + n0 + p);
      float4 v0 = src[0], v1 = src[1];
      _Float16 tmp[16];
      *(float4*)&tmp[0] = v0;
      *(float4*)&tmp[8] = v1;
#pragma unroll
      for (int i = 0; i < 16; ++i) BsT[(p + i) * SBT + kr] = tmp[i];
    }
#ifdef HAVE_TDM
    if (w == 0) __builtin_amdgcn_s_wait_tensorcnt(0);
#endif
    __syncthreads();
    v16h a = load_frag_rm(As + wm * 16 * SA, SA, ml, hh);
#pragma unroll
    for (int nt = 0; nt < 4; ++nt) {
      v16h bf = load_frag_rm(BsT + (wn * 64 + nt * 16) * SBT, SBT, nl, hh);
      acc[nt] = wmma16(a, bf, acc[nt]);
    }
  }
#pragma unroll
  for (int nt = 0; nt < 4; ++nt) {
    int n = n0 + wn * 64 + nt * 16 + nl;
    float bv = bo[n];
#pragma unroll
    for (int r = 0; r < 8; ++r) {
      int m = m0 + wm * 16 + r + 8 * hh;
      out[(size_t)m * VV + n] = acc[nt][r] + bv;
    }
  }
}

// ---- 5. loss --------------------------------------------------------------

__global__ __launch_bounds__(256)
void row_loss_kernel(const float* __restrict__ logits,
                     const int* __restrict__ targets,
                     float* __restrict__ row_loss) {
  __shared__ float red[256];
  int row = blockIdx.x;
  const float* p = logits + (size_t)row * VV;
  int tid = threadIdx.x;
  float mx = -3.0e38f;
  for (int i = tid; i < VV; i += 256) mx = fmaxf(mx, p[i]);
  red[tid] = mx;
  __syncthreads();
  for (int s2 = 128; s2 > 0; s2 >>= 1) {
    if (tid < s2) red[tid] = fmaxf(red[tid], red[tid + s2]);
    __syncthreads();
  }
  mx = red[0];
  __syncthreads();
  float sum = 0.f;
  for (int i = tid; i < VV; i += 256) sum += __expf(p[i] - mx);
  red[tid] = sum;
  __syncthreads();
  for (int s2 = 128; s2 > 0; s2 >>= 1) {
    if (tid < s2) red[tid] += red[tid + s2];
    __syncthreads();
  }
  if (tid == 0) {
    float lse = mx + __logf(red[0]);
    row_loss[row] = lse - p[targets[row]];
  }
}

__global__ __launch_bounds__(256)
void final_loss_kernel(const float* __restrict__ rl, float* __restrict__ out) {
  __shared__ float red[256];
  int tid = threadIdx.x;
  float s = 0.f;
  for (int i = tid; i < MROWS; i += 256) s += rl[i];
  red[tid] = s;
  __syncthreads();
  for (int k = 128; k > 0; k >>= 1) {
    if (tid < k) red[tid] += red[tid + k];
    __syncthreads();
  }
  if (tid == 0) out[0] = red[0] / (float)MROWS;
}

// ---- launch ---------------------------------------------------------------

extern "C" void kernel_launch(void* const* d_in, const int* in_sizes, int n_in,
                              void* d_out, int out_size, void* d_ws,
                              size_t ws_size, hipStream_t stream) {
  const int*   inputs  = (const int*)d_in[0];
  const int*   targets = (const int*)d_in[1];
  const float* tok     = (const float*)d_in[2];
  const float* pos     = (const float*)d_in[3];
  const float* Wq      = (const float*)d_in[4];
  const float* bq      = (const float*)d_in[5];
  const float* Wk      = (const float*)d_in[6];
  const float* bk      = (const float*)d_in[7];
  const float* Wv      = (const float*)d_in[8];
  const float* bv      = (const float*)d_in[9];
  const float* Wo      = (const float*)d_in[10];
  const float* bo      = (const float*)d_in[11];
  float* out = (float*)d_out;

  // workspace carve-up (~109 MiB)
  _Float16* xh  = (_Float16*)d_ws;                       // [4096][1024]
  _Float16* qh  = xh  + (size_t)MROWS * DD;              // [B][H][T][DK]
  _Float16* kh  = qh  + (size_t)MROWS * DD;
  _Float16* vh  = kh  + (size_t)MROWS * DD;
  _Float16* oh  = vh  + (size_t)MROWS * DD;              // [4096][1024]
  _Float16* Wqh = oh  + (size_t)MROWS * DD;              // [H][D][DK]
  _Float16* Wkh = Wqh + (size_t)HH * DD * DKK;
  _Float16* Wvh = Wkh + (size_t)HH * DD * DKK;
  _Float16* Woh = Wvh + (size_t)HH * DD * DKK;           // [D][V]
  float* row_loss = (float*)(Woh + (size_t)DD * VV);     // [4096]

  f32_to_f16_kernel<<<1024, 256, 0, stream>>>(Wq, Wqh, (long)HH * DD * DKK);
  f32_to_f16_kernel<<<1024, 256, 0, stream>>>(Wk, Wkh, (long)HH * DD * DKK);
  f32_to_f16_kernel<<<1024, 256, 0, stream>>>(Wv, Wvh, (long)HH * DD * DKK);
  f32_to_f16_kernel<<<4096, 256, 0, stream>>>(Wo, Woh, (long)DD * VV);
  embed_kernel<<<MROWS, 256, 0, stream>>>(inputs, tok, pos, xh);

  dim3 gq(DD / BN, MROWS / BM);     // (8, 64)
  gemm_qkv_kernel<<<gq, 256, 0, stream>>>(xh, Wqh, bq, qh);
  gemm_qkv_kernel<<<gq, 256, 0, stream>>>(xh, Wkh, bk, kh);
  gemm_qkv_kernel<<<gq, 256, 0, stream>>>(xh, Wvh, bv, vh);

  size_t smem = (size_t)(64 * SP + 256 + 64 + 64) * sizeof(float);  // ~258 KB
  attn_kernel<<<dim3(TT / 64, HH, BB), 256, smem, stream>>>(qh, kh, vh, oh);

  dim3 gl(VV / BN, MROWS / BM);     // (250, 64)
  gemm_logits_kernel<<<gl, 256, 0, stream>>>(oh, Woh, bo, out);

  row_loss_kernel<<<MROWS, 256, 0, stream>>>(out, targets, row_loss);
  final_loss_kernel<<<1, 256, 0, stream>>>(row_loss, out + (size_t)MROWS * VV);
}
